// SimpleRNN_24086176596246
// MI455X (gfx1250) — matrix operand order repaired and checked by reading, not compiled, not run on hardware
//
#include <hip/hip_runtime.h>

// ---------------------------------------------------------------------------
// Elman RNN on gfx1250 (MI455X), wave32 + v_wmma_f32_16x16x32_bf16
//   B=64, T=2048, D=H=256
//   phase 1: px = xs @ W_ih   (bandwidth-bound bf16 WMMA GEMM, ~11us floor)
//   phase 2: sequential scan  (latency-bound; dual WMMA chains, double-
//                              buffered h in LDS -> 1 barrier/step,
//                              W_hh resident in VGPRs, px prefetch)
// ---------------------------------------------------------------------------

typedef __attribute__((ext_vector_type(16))) __bf16        vbf16;
typedef __attribute__((ext_vector_type(8)))  float         v8f;
typedef __attribute__((ext_vector_type(8)))  unsigned int  v8u;

#define WMMA_BF16(A, Bm, C)                                               \
  __builtin_amdgcn_wmma_f32_16x16x32_bf16(                                \
      false, __builtin_bit_cast(vbf16, (A)),                              \
      false, __builtin_bit_cast(vbf16, (Bm)),                             \
      (short)0, (C), false, false)

static __device__ __forceinline__ unsigned short f2bf(float f) {
  unsigned int x = __float_as_uint(f);
  x += 0x7fffu + ((x >> 16) & 1u);   // round-to-nearest-even
  return (unsigned short)(x >> 16);
}

static __device__ __forceinline__ float fast_tanh(float x) {
#if __has_builtin(__builtin_amdgcn_tanhf)
  return __builtin_amdgcn_tanhf(x);   // v_tanh_f32 (gfx1250 trans op)
#else
  return tanhf(x);
#endif
}

// ---------------------------------------------------------------------------
// Kernel 0: transpose + convert a 256x256 fp32 weight into bf16, K-contiguous
// per output column: Wt[n*256 + k] = bf16(W[k*256 + n]).  (B-fragment loads
// need the two packed K values contiguous in memory.)
// ---------------------------------------------------------------------------
__global__ void prep_wt(const float* __restrict__ W,
                        unsigned short* __restrict__ Wt) {
  const int n = blockIdx.x;    // 0..255 : column of W  (N index)
  const int k = threadIdx.x;   // 0..255 : row of W     (K index)
  Wt[n * 256 + k] = f2bf(W[k * 256 + n]);
}

// ---------------------------------------------------------------------------
// Kernel 1: px[M=131072, 256] = xs[M,256] @ W_ih[256,256]
// grid = M/64 workgroups of 512 threads (16 waves).
// Wave w: mbase=(w&3)*16, nbase=(w>>2)*64 -> 4 accumulator tiles, A reused.
// xs tile (64x256 fp32) staged+converted to bf16 in LDS (32 KB).
// ---------------------------------------------------------------------------
__global__ __launch_bounds__(512) void px_gemm(
    const float* __restrict__ xs, const unsigned short* __restrict__ WtI,
    float* __restrict__ px) {
  __shared__ unsigned short xt[64 * 256];

  const int tid = threadIdx.x;
  const size_t row0 = (size_t)blockIdx.x * 64;

  const float* src = xs + row0 * 256;
  for (int i = tid; i < 64 * 256; i += 512) xt[i] = f2bf(src[i]);
  __syncthreads();

  const int w = tid >> 5, lane = tid & 31;
  const int l15 = lane & 15, hi = lane >> 4;
  const int mbase = (w & 3) * 16;
  const int nbase = (w >> 2) * 64;

  v8f acc[4] = {};
  for (int kb = 0; kb < 256; kb += 32) {
    // A fragment: 16x32 bf16, row m = mbase+l15, K pairs per ISA layout table
    v8u a;
#pragma unroll
    for (int v = 0; v < 8; ++v) {
      const int k = kb + (v < 4 ? 2 * v : 2 * v + 8) + 8 * hi;
      a[v] = *(const unsigned int*)&xt[(mbase + l15) * 256 + k];
    }
#pragma unroll
    for (int nt = 0; nt < 4; ++nt) {
      // B fragment: 32x16 bf16 from transposed weights (K contiguous)
      v8u b;
#pragma unroll
      for (int v = 0; v < 8; ++v) {
        const int k = kb + (v < 4 ? 2 * v : 2 * v + 8) + 8 * hi;
        b[v] = *(const unsigned int*)&WtI[(size_t)(nbase + nt * 16 + l15) * 256 + k];
      }
      acc[nt] = WMMA_BF16(a, b, acc[nt]);
    }
  }

  // C/D layout: VGPR r -> M = r + 8*hi, N = l15
#pragma unroll
  for (int nt = 0; nt < 4; ++nt) {
#pragma unroll
    for (int r = 0; r < 8; ++r) {
      const size_t row = row0 + mbase + r + 8 * hi;
      px[row * 256 + nbase + nt * 16 + l15] = acc[nt][r];
    }
  }
}

// ---------------------------------------------------------------------------
// Kernel 2: sequential scan.  4 workgroups (one 16-row batch tile each),
// 16 waves x 32 lanes.  Wave w owns output columns [16w, 16w+16); its eight
// 32x16 B fragments of W_hh stay resident in VGPRs across all 2048 steps.
// h (16x256 bf16) double-buffered in LDS -> one barrier per step.
// Two independent accumulator chains (K even/odd blocks) halve the
// dependent-WMMA critical path; px for t is loaded at step top, px for t+1
// is prefetched (global_prefetch_b8).
// ---------------------------------------------------------------------------
__global__ __launch_bounds__(512) void rnn_scan(
    const float* __restrict__ c0, const float* __restrict__ bias,
    const unsigned short* __restrict__ WtH, const float* __restrict__ px,
    float* __restrict__ hfin, float* __restrict__ ys) {
  __shared__ unsigned short hs[2][16 * 256];

  const int tid = threadIdx.x;
  const int wgb = blockIdx.x;  // batch tile 0..3
  const int w = tid >> 5, lane = tid & 31;
  const int l15 = lane & 15, hi = lane >> 4;
  const int nbase = w * 16;
  const int col = nbase + l15;

  // init h_0 from carry c0 into buffer 0
  for (int i = tid; i < 16 * 256; i += 512) {
    const int m = i >> 8, n = i & 255;
    hs[0][i] = f2bf(c0[(size_t)(wgb * 16 + m) * 256 + n]);
  }

  // resident W_hh B-fragments: 8 K-steps x 8 dwords = 64 VGPRs
  v8u Bf[8];
#pragma unroll
  for (int ks = 0; ks < 8; ++ks) {
#pragma unroll
    for (int v = 0; v < 8; ++v) {
      const int k = ks * 32 + (v < 4 ? 2 * v : 2 * v + 8) + 8 * hi;
      Bf[ks][v] = *(const unsigned int*)&WtH[(size_t)(nbase + l15) * 256 + k];
    }
  }
  const float bv = bias[col];

  // per-lane px/ys row pointers for the 8 C rows this lane owns
  const float* pxp[8];
  float*       ysp[8];
#pragma unroll
  for (int r = 0; r < 8; ++r) {
    const size_t b = (size_t)wgb * 16 + r + 8 * hi;
    pxp[r] = px + (b * 2048) * 256 + col;
    ysp[r] = ys + (b * 2048) * 256 + col;
  }

  __syncthreads();

  for (int t = 0; t < 2048; ++t) {
    const int cur = t & 1, nxt = cur ^ 1;
    const unsigned short* hcur = hs[cur];

    // px for this step (issued first; scheduler hoists across WMMA chain),
    // plus prefetch of next step's rows (+1 KB stride).
    float pre[8];
#pragma unroll
    for (int r = 0; r < 8; ++r) {
      pre[r] = pxp[r][(size_t)t * 256];
      __builtin_prefetch(&pxp[r][(size_t)(t + 1) * 256], 0, 0);
    }

    // two independent accumulator chains over K blocks (even/odd)
    v8f acc0 = {}, acc1 = {};
#pragma unroll
    for (int ks = 0; ks < 4; ++ks) {
      v8u a0, a1;
#pragma unroll
      for (int v = 0; v < 8; ++v) {
        const int koff = (v < 4 ? 2 * v : 2 * v + 8) + 8 * hi;
        a0[v] = *(const unsigned int*)&hcur[l15 * 256 + (2 * ks) * 32 + koff];
        a1[v] = *(const unsigned int*)&hcur[l15 * 256 + (2 * ks + 1) * 32 + koff];
      }
      acc0 = WMMA_BF16(a0, Bf[2 * ks], acc0);
      acc1 = WMMA_BF16(a1, Bf[2 * ks + 1], acc1);
    }

    float out[8];
#pragma unroll
    for (int r = 0; r < 8; ++r) {
      const float val = fast_tanh(acc0[r] + acc1[r] + bv + pre[r]);
      out[r] = val;
      ysp[r][(size_t)t * 256] = val;
      hs[nxt][(r + 8 * hi) * 256 + col] = f2bf(val);
    }
    if (t == 2047) {
#pragma unroll
      for (int r = 0; r < 8; ++r)
        hfin[(size_t)(wgb * 16 + r + 8 * hi) * 256 + col] = out[r];
    }
    // single barrier: orders writes(hs[nxt]) before next step's reads, and
    // this step's reads(hs[cur]) before step t+1 overwrites it.
    __syncthreads();
  }
}

// ---------------------------------------------------------------------------
extern "C" void kernel_launch(void* const* d_in, const int* in_sizes, int n_in,
                              void* d_out, int out_size, void* d_ws,
                              size_t ws_size, hipStream_t stream) {
  (void)in_sizes; (void)n_in; (void)out_size; (void)ws_size;

  // setup_inputs order: c, xs, W_ih, W_hh, b_hh
  const float* c   = (const float*)d_in[0];
  const float* xs  = (const float*)d_in[1];
  const float* Wih = (const float*)d_in[2];
  const float* Whh = (const float*)d_in[3];
  const float* bhh = (const float*)d_in[4];

  float* hfin = (float*)d_out;                 // [64,256]
  float* ys   = (float*)d_out + 64 * 256;      // [64,2048,256]

  // workspace: px fp32 (128 MB) | Wt_ih bf16 (128 KB) | Wt_hh bf16 (128 KB)
  char* ws = (char*)d_ws;
  float* px = (float*)ws;
  const size_t px_bytes = (size_t)64 * 2048 * 256 * sizeof(float);
  unsigned short* WtI = (unsigned short*)(ws + px_bytes);
  unsigned short* WtH = WtI + 256 * 256;

  prep_wt<<<256, 256, 0, stream>>>(Wih, WtI);
  prep_wt<<<256, 256, 0, stream>>>(Whh, WtH);
  px_gemm<<<131072 / 64, 512, 0, stream>>>(xs, WtI, px);
  rnn_scan<<<4, 512, 0, stream>>>(c, bhh, WtH, px, hfin, ys);
}
